// Attention_talking_head_50843822850672
// MI455X (gfx1250) — matrix operand order repaired and checked by reading, not compiled, not run on hardware
//
#include <hip/hip_runtime.h>
#include <hip/hip_bf16.h>

typedef __attribute__((ext_vector_type(16))) __bf16     v16bf;
typedef __attribute__((ext_vector_type(8)))  float      v8f;
typedef __attribute__((ext_vector_type(4)))  unsigned   u32x4;
typedef __attribute__((ext_vector_type(4)))  unsigned short u16x4;
typedef unsigned short ushort_t;

#define DIM      768
#define HEADS    12
#define HEAD_DIM 64
#define NB       4
#define NSEQ     1024
#define ROWS     (NB * NSEQ)      // 4096
#define QKV_COLS (3 * DIM)        // 2304
#define SCALE_Q  0.125f           // 64^-0.5

// ---------- bf16 helpers ----------
__device__ __forceinline__ ushort_t f2bf(float f) {
    unsigned u = __float_as_uint(f);
    unsigned r = u + 0x7FFFu + ((u >> 16) & 1u);
    return (ushort_t)(r >> 16);
}
__device__ __forceinline__ float bf2f(ushort_t h) {
    return __uint_as_float(((unsigned)h) << 16);
}

// ---------- WMMA fragment loaders (bf16, 16x16x32) ----------
// A (16xK row-major, stored A[row][k]): lane L: row = L&15, two 8-half runs
// at k = ko and k = ko+16 with ko = (L>>4)*8.
__device__ __forceinline__ v16bf load_frag_A(const ushort_t* base, int ld, int row0, int k0) {
    int lane = threadIdx.x & 31;
    const ushort_t* p = base + (size_t)(row0 + (lane & 15)) * ld + k0 + ((lane >> 4) << 3);
    union { u32x4 u[2]; v16bf v; } U;
    U.u[0] = *(const u32x4*)(p);
    U.u[1] = *(const u32x4*)(p + 16);
    return U.v;
}
// B (KxN, sourced from Bsrc[col][k] row-major i.e. Bsrc = B^T): lane L:
// col = L&15, two 8-half runs at k = ko and k = ko+16 with ko = (L>>4)*8.
__device__ __forceinline__ v16bf load_frag_B(const ushort_t* base, int ld, int col0, int k0) {
    int lane = threadIdx.x & 31;
    const ushort_t* p = base + (size_t)(col0 + (lane & 15)) * ld + k0 + ((lane >> 4) << 3);
    union { u32x4 u[2]; v16bf v; } U;
    U.u[0] = *(const u32x4*)(p);
    U.u[1] = *(const u32x4*)(p + 16);
    return U.v;
}

// one wave computes a 32x64 tile: two A fragments, each B fragment feeds two
// WMMAs (1.5 b128-loads per WMMA).  acc[t] = rows 0-15, acc[4+t] = rows 16-31.
__device__ __forceinline__ void gemm_tile_32x64(const ushort_t* Ab, int lda, int row0,
                                                const ushort_t* Bb, int ldb, int col0,
                                                int K, v8f acc[8]) {
    for (int kb = 0; kb < K; kb += 32) {
        v16bf a0 = load_frag_A(Ab, lda, row0, kb);
        v16bf a1 = load_frag_A(Ab, lda, row0 + 16, kb);
#pragma unroll
        for (int t = 0; t < 4; ++t) {
            v16bf b = load_frag_B(Bb, ldb, col0 + 16 * t, kb);
            acc[t]     = __builtin_amdgcn_wmma_f32_16x16x32_bf16(
                false, a0, false, b, (short)0, acc[t],     false, false);
            acc[4 + t] = __builtin_amdgcn_wmma_f32_16x16x32_bf16(
                false, a1, false, b, (short)0, acc[4 + t], false, false);
        }
    }
}

// ---------- kernels ----------
__global__ void __launch_bounds__(256) k_pack(const float* a, ushort_t* o, int n) {
    int i = blockIdx.x * blockDim.x + threadIdx.x;
    int stride = gridDim.x * blockDim.x;
    for (; i < n; i += stride) o[i] = f2bf(a[i]);
}

// QKV GEMM: [4096,768] x [768,2304] -> split into q(scaled)/k/v, bf16 [B,H,N,64]
__global__ void __launch_bounds__(256) k_qkv(const ushort_t* xb, const ushort_t* wq,
                                             ushort_t* q, ushort_t* kk, ushort_t* v) {
    int w = blockIdx.x * 8 + (threadIdx.x >> 5);     // 4608 waves
    int mtile = w / 36, ngrp = w % 36;               // mtile: 32-row blocks
    int lane = threadIdx.x & 31;
    v8f acc[8] = {};
    gemm_tile_32x64(xb, DIM, mtile * 32, wq, DIM, ngrp * 64, DIM, acc);

    int which = ngrp / 12, h = ngrp % 12;
    int lrow = (lane >> 4) << 3, lc = lane & 15;
    ushort_t* dst = (which == 0) ? q : (which == 1) ? kk : v;
    float scl = (which == 0) ? SCALE_Q : 1.0f;
#pragma unroll
    for (int a = 0; a < 2; ++a)
#pragma unroll
        for (int t = 0; t < 4; ++t)
#pragma unroll
            for (int e = 0; e < 8; ++e) {
                int i = mtile * 32 + a * 16 + e + lrow;
                int b = i >> 10, n = i & 1023;
                int d = t * 16 + lc;
                dst[(((size_t)(b * 12 + h) * NSEQ + n) * HEAD_DIM) + d] =
                    f2bf(acc[a * 4 + t][e] * scl);
            }
}

// v [B,H,N,64] -> vt [B,H,64,N] via LDS tiles
__global__ void __launch_bounds__(256) k_transpose_v(const ushort_t* v, ushort_t* vt) {
    __shared__ ushort_t tile[64][65];
    int bh = blockIdx.x >> 4, mblk = blockIdx.x & 15;
    const ushort_t* src = v  + (size_t)bh * NSEQ * HEAD_DIM;
    ushort_t*       dst = vt + (size_t)bh * HEAD_DIM * NSEQ;
    int c = threadIdx.x & 63, r4 = threadIdx.x >> 6;
#pragma unroll
    for (int it = 0; it < 16; ++it) {
        int m = it * 4 + r4;
        tile[m][c] = src[(size_t)(mblk * 64 + m) * HEAD_DIM + c];
    }
    __syncthreads();
#pragma unroll
    for (int it = 0; it < 16; ++it) {
        int d = it * 4 + r4;
        dst[(size_t)d * NSEQ + mblk * 64 + c] = tile[c][d];
    }
}

// scores: raw QK^T for 12 heads per (b, 16x16 n/m tile), W_l head-mix, store bf16
__global__ void __launch_bounds__(256) k_scores(const ushort_t* q, const ushort_t* kk,
                                                ushort_t* S, const float* Wl, const float* bl) {
    int w = blockIdx.x * 8 + (threadIdx.x >> 5);     // 16384 waves
    int b = w >> 12, rest = w & 4095;
    int nt = rest >> 6, mt = rest & 63;
    int lane = threadIdx.x & 31;
    v8f acc[12] = {};
#pragma unroll
    for (int h = 0; h < 12; ++h) {
        const ushort_t* qb = q  + (size_t)(b * 12 + h) * NSEQ * HEAD_DIM;
        const ushort_t* kb = kk + (size_t)(b * 12 + h) * NSEQ * HEAD_DIM;
#pragma unroll
        for (int k0 = 0; k0 < HEAD_DIM; k0 += 32) {
            v16bf a  = load_frag_A(qb, HEAD_DIM, nt * 16, k0);
            v16bf bf = load_frag_B(kb, HEAD_DIM, mt * 16, k0);
            acc[h] = __builtin_amdgcn_wmma_f32_16x16x32_bf16(
                false, a, false, bf, (short)0, acc[h], false, false);
        }
    }
    int lrow = (lane >> 4) << 3, lc = lane & 15;
#pragma unroll
    for (int g = 0; g < 12; ++g) {
        float se[8];
#pragma unroll
        for (int e = 0; e < 8; ++e) se[e] = bl[g];
#pragma unroll
        for (int h = 0; h < 12; ++h) {
            float wv = Wl[g * 12 + h];
#pragma unroll
            for (int e = 0; e < 8; ++e) se[e] += wv * acc[h][e];
        }
        ushort_t* row = S + ((size_t)(b * 12 + g) * NSEQ + nt * 16) * NSEQ + mt * 16 + lc;
#pragma unroll
        for (int e = 0; e < 8; ++e) row[(size_t)(e + lrow) * NSEQ] = f2bf(se[e]);
    }
}

// fused softmax (over m) + W_w mix + lambda rescale, in place over S.
// one block per (b,n); thread t owns m = 4t..4t+3 for all 12 heads.
__global__ void __launch_bounds__(256) k_softmax_mix(ushort_t* S, const float* Ww,
                                                     const float* bw, const float* lamb) {
    __shared__ float red[12][8];
    __shared__ float gmax[12], gsum[12];
    int b = blockIdx.x >> 10, n = blockIdx.x & 1023;
    int t = threadIdx.x, lane = t & 31, wv = t >> 5;
    int m0 = t * 4;
    float p[12][4];
#pragma unroll
    for (int g = 0; g < 12; ++g) {
        const ushort_t* row = S + ((size_t)(b * 12 + g) * NSEQ + n) * NSEQ;
        u16x4 u = *(const u16x4*)(row + m0);
#pragma unroll
        for (int j = 0; j < 4; ++j) p[g][j] = bf2f(u[j]);
    }
    // row max
    float lm[12];
#pragma unroll
    for (int g = 0; g < 12; ++g) {
        float m = fmaxf(fmaxf(p[g][0], p[g][1]), fmaxf(p[g][2], p[g][3]));
#pragma unroll
        for (int off = 16; off > 0; off >>= 1) m = fmaxf(m, __shfl_xor(m, off, 32));
        lm[g] = m;
    }
    if (lane == 0)
#pragma unroll
        for (int g = 0; g < 12; ++g) red[g][wv] = lm[g];
    __syncthreads();
    if (t < 12) {
        float m = red[t][0];
#pragma unroll
        for (int w2 = 1; w2 < 8; ++w2) m = fmaxf(m, red[t][w2]);
        gmax[t] = m;
    }
    __syncthreads();
    // exp + row sum
    float ls[12];
#pragma unroll
    for (int g = 0; g < 12; ++g) {
        float gm = gmax[g], s = 0.f;
#pragma unroll
        for (int j = 0; j < 4; ++j) { p[g][j] = __expf(p[g][j] - gm); s += p[g][j]; }
#pragma unroll
        for (int off = 16; off > 0; off >>= 1) s += __shfl_xor(s, off, 32);
        ls[g] = s;
    }
    __syncthreads();
    if (lane == 0)
#pragma unroll
        for (int g = 0; g < 12; ++g) red[g][wv] = ls[g];
    __syncthreads();
    if (t < 12) {
        float s = 0.f;
#pragma unroll
        for (int w2 = 0; w2 < 8; ++w2) s += red[t][w2];
        gsum[t] = s;
    }
    __syncthreads();
#pragma unroll
    for (int g = 0; g < 12; ++g) {
        float r = 1.0f / gsum[g];
#pragma unroll
        for (int j = 0; j < 4; ++j) p[g][j] *= r;
    }
    // post mix:  F = (1+lam)*(Ww . p) + (1+lam)*bw - lam/N   (in place; each
    // thread only rewrites the exact positions it loaded)
#pragma unroll
    for (int gp = 0; gp < 12; ++gp) {
        float c1 = 1.0f + lamb[gp];
        float c0 = c1 * bw[gp] - lamb[gp] * (1.0f / (float)NSEQ);
        float o[4] = {c0, c0, c0, c0};
#pragma unroll
        for (int g = 0; g < 12; ++g) {
            float wv2 = c1 * Ww[gp * 12 + g];
#pragma unroll
            for (int j = 0; j < 4; ++j) o[j] += wv2 * p[g][j];
        }
        ushort_t* row = S + ((size_t)(b * 12 + gp) * NSEQ + n) * NSEQ;
        u16x4 u;
#pragma unroll
        for (int j = 0; j < 4; ++j) u[j] = f2bf(o[j]);
        *(u16x4*)(row + m0) = u;
    }
}

// AV GEMM per (b,g): [1024x1024] x vt^T -> Y[b,n, g*64+d] bf16
__global__ void __launch_bounds__(256) k_av(const ushort_t* S, const ushort_t* vt, ushort_t* Y) {
    int w = blockIdx.x * 8 + (threadIdx.x >> 5);     // 1536 waves
    int b = w / 384, r = w % 384;                    // 384 = 12 heads * 32 ntiles
    int g = r >> 5, nt = r & 31;                     // nt: 32-row blocks
    int lane = threadIdx.x & 31;
    const ushort_t* A  = S  + (size_t)(b * 12 + g) * NSEQ * NSEQ;
    const ushort_t* Bb = vt + (size_t)(b * 12 + g) * HEAD_DIM * NSEQ;
    v8f acc[8] = {};
    gemm_tile_32x64(A, NSEQ, nt * 32, Bb, NSEQ, 0, NSEQ, acc);
    int lrow = (lane >> 4) << 3, lc = lane & 15;
#pragma unroll
    for (int a = 0; a < 2; ++a)
#pragma unroll
        for (int t = 0; t < 4; ++t)
#pragma unroll
            for (int e = 0; e < 8; ++e) {
                int n = nt * 32 + a * 16 + e + lrow;
                int d = t * 16 + lc;
                Y[((size_t)(b * NSEQ + n)) * DIM + g * 64 + d] = f2bf(acc[a * 4 + t][e]);
            }
}

// output projection: [4096,768] x [768,768]^T + b_proj -> fp32 out
__global__ void __launch_bounds__(256) k_proj(const ushort_t* Yb, const ushort_t* wp,
                                              const float* bp, float* out) {
    int w = blockIdx.x * 8 + (threadIdx.x >> 5);     // 1536 waves
    int mt = w / 12, ngrp = w % 12;                  // mt: 32-row blocks
    int lane = threadIdx.x & 31;
    v8f acc[8] = {};
    gemm_tile_32x64(Yb, DIM, mt * 32, wp, DIM, ngrp * 64, DIM, acc);
    int lrow = (lane >> 4) << 3, lc = lane & 15;
#pragma unroll
    for (int a = 0; a < 2; ++a)
#pragma unroll
        for (int t = 0; t < 4; ++t)
#pragma unroll
            for (int e = 0; e < 8; ++e) {
                int i = mt * 32 + a * 16 + e + lrow;
                int j = ngrp * 64 + t * 16 + lc;
                out[(size_t)i * DIM + j] = acc[a * 4 + t][e] + bp[j];
            }
}

extern "C" void kernel_launch(void* const* d_in, const int* in_sizes, int n_in,
                              void* d_out, int out_size, void* d_ws, size_t ws_size,
                              hipStream_t stream) {
    const float* x      = (const float*)d_in[0];
    const float* W_qkv  = (const float*)d_in[1];
    const float* W_proj = (const float*)d_in[2];
    const float* b_proj = (const float*)d_in[3];
    const float* W_l    = (const float*)d_in[4];
    const float* b_l    = (const float*)d_in[5];
    const float* W_w    = (const float*)d_in[6];
    const float* b_w    = (const float*)d_in[7];
    const float* lamb   = (const float*)d_in[8];
    float* out = (float*)d_out;

    char* ws = (char*)d_ws;
    size_t off = 0;
    ushort_t* xb = (ushort_t*)(ws + off); off += (size_t)ROWS * DIM * 2;          // 6.29 MB
    ushort_t* wq = (ushort_t*)(ws + off); off += (size_t)QKV_COLS * DIM * 2;      // 3.54 MB
    ushort_t* wp = (ushort_t*)(ws + off); off += (size_t)DIM * DIM * 2;           // 1.18 MB
    ushort_t* q  = (ushort_t*)(ws + off); off += (size_t)ROWS * DIM * 2;          // 6.29 MB
    ushort_t* kk = (ushort_t*)(ws + off); off += (size_t)ROWS * DIM * 2;          // 6.29 MB
    ushort_t* v  = (ushort_t*)(ws + off); off += (size_t)ROWS * DIM * 2;          // 6.29 MB
    ushort_t* vt = (ushort_t*)(ws + off); off += (size_t)ROWS * DIM * 2;          // 6.29 MB
    ushort_t* S  = (ushort_t*)(ws + off); off += (size_t)NB * HEADS * NSEQ * NSEQ * 2; // 100.7 MB (L2-resident)
    ushort_t* Y  = (ushort_t*)(ws + off); off += (size_t)ROWS * DIM * 2;          // 6.29 MB

    k_pack<<<2048, 256, 0, stream>>>(x,      xb, ROWS * DIM);
    k_pack<<<2048, 256, 0, stream>>>(W_qkv,  wq, QKV_COLS * DIM);
    k_pack<<<1024, 256, 0, stream>>>(W_proj, wp, DIM * DIM);

    k_qkv<<<576, 256, 0, stream>>>(xb, wq, q, kk, v);            // 4608 waves, 32x64 tiles
    k_transpose_v<<<768, 256, 0, stream>>>(v, vt);
    k_scores<<<2048, 256, 0, stream>>>(q, kk, S, W_l, b_l);      // 16384 waves
    k_softmax_mix<<<4096, 256, 0, stream>>>(S, W_w, b_w, lamb);
    k_av<<<192, 256, 0, stream>>>(S, vt, Y);                     // 1536 waves, 32x64 tiles
    k_proj<<<192, 256, 0, stream>>>(Y, wp, b_proj, out);         // 1536 waves, 32x64 tiles
}
